// MaskedCrossAttentionLayer_34299608825999
// MI455X (gfx1250) — matrix operand order repaired and checked by reading, not compiled, not run on hardware
//
#include <hip/hip_runtime.h>
#include <hip/hip_bf16.h>
#include <math.h>

typedef __attribute__((ext_vector_type(16))) __bf16 v16bf;
typedef __attribute__((ext_vector_type(8)))  __bf16 v8bf;
typedef __attribute__((ext_vector_type(4)))  __bf16 v4bf;
typedef __attribute__((ext_vector_type(8)))  float  v8f;
typedef __attribute__((ext_vector_type(4)))  float  v4f;

#define BATCH 8
#define QLEN 100
#define QPAD 112
#define HW 16384
#define DMODEL 256
#define NHEAD 8
#define HDIM 32
#define BH (BATCH * NHEAD)
#define NSEG 8
#define QTILES 7               // ceil(100/16)
#define NTILES (BH * QTILES)   // 448
#define RECF 544               // 16*32 O + 16 m + 16 l floats per partial record

// ---------- operand loaders ----------

// A operand (16x32 bf16): lane L -> row L%16; elements: 16 contiguous at p0
static __device__ inline v16bf load_a_bf(const __bf16* p0) {
  v8bf lo = *(const v8bf*)p0;
  v8bf hi = *(const v8bf*)(p0 + 8);
  return __builtin_shufflevector(lo, hi, 0,1,2,3,4,5,6,7,8,9,10,11,12,13,14,15);
}

// B operand (32x16 bf16): lane L -> col L%16; 16 contiguous K elements at p
static __device__ inline v16bf load_b_bf(const __bf16* p) {
  return *(const v16bf*)p;
}

// A operand from f32 memory (convert to bf16): 16 contiguous floats at p0
static __device__ inline v16bf cvt_a_f32(const float* p0) {
  v8f lo = *(const v8f*)p0;
  v8f hi = *(const v8f*)(p0 + 8);
  v8bf blo = __builtin_convertvector(lo, v8bf);
  v8bf bhi = __builtin_convertvector(hi, v8bf);
  return __builtin_shufflevector(blo, bhi, 0,1,2,3,4,5,6,7,8,9,10,11,12,13,14,15);
}

static __device__ inline v8f wmma_bf16(v16bf a, v16bf b, v8f c) {
  return __builtin_amdgcn_wmma_f32_16x16x32_bf16(false, a, false, b, (short)0, c,
                                                 false, false);
}

// ---------- kernel 0: bulk f32 -> bf16 conversion (4 elements/thread) ----------
__global__ __launch_bounds__(256) void cvt_bf16(const float* __restrict__ src,
                                                __bf16* __restrict__ dst, int n4) {
  int i = blockIdx.x * blockDim.x + threadIdx.x;
  if (i < n4) {
    v4f f = ((const v4f*)src)[i];
    ((v4bf*)dst)[i] = __builtin_convertvector(f, v4bf);
  }
}

// ---------- kernel 1: K/V projection (bf16 in, bf16 out) ----------
// wave tile: 16 rows x 128 cols (8 accumulators). cols 0..255 -> K, 256..511 -> V.
__global__ __launch_bounds__(256) void kv_proj(const __bf16* __restrict__ kvbf,
                                               const __bf16* __restrict__ wbf,
                                               const float* __restrict__ bias,
                                               __bf16* __restrict__ kbf,
                                               __bf16* __restrict__ vT) {
  int lane = threadIdx.x & 31, wid = threadIdx.x >> 5;
  int w = blockIdx.x * 8 + wid;          // 32768 waves
  int cg = w & 3;                        // col group of 128
  int mt = w >> 2;                       // 8192 row tiles
  int half = lane >> 4, ln = lane & 15;

  int grow0 = mt * 16;
  const __bf16* arow = kvbf + (size_t)(grow0 + ln) * DMODEL + half * 16;
  const __bf16* wrow = wbf + (size_t)(256 + cg * 128 + ln) * DMODEL + half * 16;

  v8f acc[8];
#pragma unroll
  for (int ct = 0; ct < 8; ++ct) acc[ct] = (v8f){0,0,0,0,0,0,0,0};

#pragma unroll
  for (int kk = 0; kk < 8; ++kk) {
    // stage A + all 8 B tiles first so loads issue as one clause,
    // then the WMMA chain drains them with descending loadcnt waits
    v16bf A = load_a_bf(arow + kk * 32);
    v16bf Bv[8];
#pragma unroll
    for (int ct = 0; ct < 8; ++ct)
      Bv[ct] = load_b_bf(wrow + (size_t)(ct * 16) * DMODEL + kk * 32);
#pragma unroll
    for (int ct = 0; ct < 8; ++ct) acc[ct] = wmma_bf16(A, Bv[ct], acc[ct]);
  }

  int b = grow0 >> 14;            // /HW
  int keyb = grow0 & (HW - 1);
#pragma unroll
  for (int ct = 0; ct < 8; ++ct) {
    int c = cg * 128 + ct * 16 + ln;
    float bb = bias[256 + c];
    if (c < 256) {  // K -> [bh][key][32]
      int h = c >> 5, dd = c & 31;
      __bf16* kout = kbf + ((size_t)(b * NHEAD + h) * HW) * HDIM + dd;
#pragma unroll
      for (int r = 0; r < 8; ++r) {
        int key = keyb + r + 8 * half;
        kout[(size_t)key * HDIM] = (__bf16)(acc[ct][r] + bb);
      }
    } else {        // V -> vT[bh][d][HW], 8 consecutive keys per lane -> b128
      int cv = c - 256;
      int h = cv >> 5, dd = cv & 31;
      v8f av = acc[ct] + bb;
      v8bf pk = __builtin_convertvector(av, v8bf);
      __bf16* vp = vT + ((size_t)(b * NHEAD + h) * HDIM + dd) * HW + keyb + 8 * half;
      *(v8bf*)vp = pk;
    }
  }
}

// ---------- kernel 2: Q projection (pre-scaled by 1/sqrt(32)) ----------
__global__ __launch_bounds__(256) void q_proj(const float* __restrict__ query,
                                              const __bf16* __restrict__ wbf,
                                              const float* __restrict__ bias,
                                              __bf16* __restrict__ qbf) {
  int lane = threadIdx.x & 31, wid = threadIdx.x >> 5;
  int w = blockIdx.x * 8 + wid;           // 224 waves exactly
  int cg = w & 3;
  int t = w >> 2;                         // 56 = 8 b * 7 qtiles
  int b = t / QTILES, qt = t % QTILES;
  int half = lane >> 4, ln = lane & 15;

  int srow = qt * 16 + ln; if (srow > QLEN - 1) srow = QLEN - 1;
  const float* arow = query + ((size_t)b * QLEN + srow) * DMODEL + half * 16;
  const __bf16* wrow = wbf + (size_t)(cg * 64 + ln) * DMODEL + half * 16;

  v8f acc[4];
#pragma unroll
  for (int ct = 0; ct < 4; ++ct) acc[ct] = (v8f){0,0,0,0,0,0,0,0};

#pragma unroll
  for (int kk = 0; kk < 8; ++kk) {
    v16bf A = cvt_a_f32(arow + kk * 32);
    v16bf Bv[4];
#pragma unroll
    for (int ct = 0; ct < 4; ++ct)
      Bv[ct] = load_b_bf(wrow + (size_t)(ct * 16) * DMODEL + kk * 32);
#pragma unroll
    for (int ct = 0; ct < 4; ++ct) acc[ct] = wmma_bf16(A, Bv[ct], acc[ct]);
  }

  const float scale = 0.17677669529663687f;  // 1/sqrt(32)
#pragma unroll
  for (int ct = 0; ct < 4; ++ct) {
    int c = cg * 64 + ct * 16 + ln;
    int h = c >> 5, dd = c & 31;
    float bb = bias[c];
#pragma unroll
    for (int r = 0; r < 8; ++r) {
      int qrow = qt * 16 + r + 8 * half;   // store all 112 padded rows
      qbf[((size_t)(b * NHEAD + h) * QPAD + qrow) * HDIM + dd] =
          (__bf16)((acc[ct][r] + bb) * scale);
    }
  }
}

// ---------- kernel 3: flash attention partials ----------
__global__ __launch_bounds__(256) void attn_partial(const __bf16* __restrict__ qbf,
                                                    const __bf16* __restrict__ kbf,
                                                    const __bf16* __restrict__ vT,
                                                    const unsigned char* __restrict__ mask,
                                                    float* __restrict__ part) {
  __shared__ __bf16 plds[8][16 * 32];
  int lane = threadIdx.x & 31, wid = threadIdx.x >> 5;
  int w = blockIdx.x * 8 + wid;        // 3584 waves
  int seg = w & (NSEG - 1);
  int t = w >> 3;                      // 448 tiles
  int bh = t / QTILES, qt = t % QTILES;
  int half = lane >> 4, ln = lane & 15;

  // q A-operand (persistent)
  const __bf16* qrow =
      qbf + ((size_t)bh * QPAD + qt * 16 + ln) * HDIM + half * 16;
  v16bf qa = load_a_bf(qrow);

  const __bf16* kbase = kbf + (size_t)bh * HW * HDIM;
  const __bf16* vbase = vT + (size_t)bh * HDIM * HW;
  const unsigned char* mb = mask + (size_t)bh * QLEN * HW;
  int moff[8];
#pragma unroll
  for (int r = 0; r < 8; ++r) {
    int row = qt * 16 + r + 8 * half;
    moff[r] = (row < QLEN ? row : QLEN - 1) * HW;
  }

  float m[8], l[8];
  v8f o0 = (v8f){0,0,0,0,0,0,0,0}, o1 = (v8f){0,0,0,0,0,0,0,0};
#pragma unroll
  for (int r = 0; r < 8; ++r) { m[r] = -INFINITY; l[r] = 0.f; }

  __bf16* tile = plds[wid];
  int k0 = seg * (HW / NSEG);

  for (int kb = k0; kb < k0 + HW / NSEG; kb += 32) {
    // issue K and V operand loads together: V is consumed only after the
    // softmax VALU block, so its latency is hidden by ~200 VALU ops
    v16bf bk0 = load_b_bf(kbase + (size_t)(kb + ln) * HDIM + half * 16);
    v16bf bk1 = load_b_bf(kbase + (size_t)(kb + 16 + ln) * HDIM + half * 16);
    v16bf bv0 = load_b_bf(vbase + (size_t)ln * HW + kb + half * 16);
    v16bf bv1 = load_b_bf(vbase + (size_t)(16 + ln) * HW + kb + half * 16);

    v8f z = (v8f){0,0,0,0,0,0,0,0};
    v8f s0 = wmma_bf16(qa, bk0, z);
    v8f s1 = wmma_bf16(qa, bk1, z);

    float t0[8], t1[8], cm[8];
#pragma unroll
    for (int r = 0; r < 8; ++r) {
      unsigned char m0 = mb[moff[r] + kb + ln];
      unsigned char m1 = mb[moff[r] + kb + 16 + ln];
      t0[r] = m0 ? s0[r] : -1e30f;
      t1[r] = m1 ? s1[r] : -1e30f;
      cm[r] = fmaxf(t0[r], t1[r]);
    }
    // row max across the 16 lanes of each half
#pragma unroll
    for (int r = 0; r < 8; ++r)
#pragma unroll
      for (int sh = 1; sh < 16; sh <<= 1)
        cm[r] = fmaxf(cm[r], __shfl_xor(cm[r], sh, 32));

    float p0[8], p1[8], rs[8];
#pragma unroll
    for (int r = 0; r < 8; ++r) {
      float mn = fmaxf(m[r], cm[r]);
      float al = __expf(m[r] - mn);
      m[r] = mn;
      p0[r] = __expf(t0[r] - mn);
      p1[r] = __expf(t1[r] - mn);
      rs[r] = p0[r] + p1[r];
      l[r] *= al;
      o0[r] *= al;
      o1[r] *= al;
    }
#pragma unroll
    for (int r = 0; r < 8; ++r) {
#pragma unroll
      for (int sh = 1; sh < 16; sh <<= 1) rs[r] += __shfl_xor(rs[r], sh, 32);
      l[r] += rs[r];
    }

    // probs: C-layout -> A-operand layout via LDS
#pragma unroll
    for (int r = 0; r < 8; ++r) {
      int row = r + 8 * half;
      tile[row * 32 + ln] = (__bf16)p0[r];
      tile[row * 32 + 16 + ln] = (__bf16)p1[r];
    }
    asm volatile("s_wait_dscnt 0" ::: "memory");
    v16bf pa = load_a_bf(tile + ln * 32 + half * 16);

    o0 = wmma_bf16(pa, bv0, o0);
    o1 = wmma_bf16(pa, bv1, o1);
  }

  float* rec = part + ((size_t)t * NSEG + seg) * RECF;
#pragma unroll
  for (int r = 0; r < 8; ++r) {
    int row = r + 8 * half;
    rec[row * 32 + ln] = o0[r];
    rec[row * 32 + 16 + ln] = o1[r];
  }
  if (ln == 0) {
#pragma unroll
    for (int r = 0; r < 8; ++r) {
      int row = r + 8 * half;
      rec[512 + row] = m[r];
      rec[528 + row] = l[r];
    }
  }
}

// ---------- kernel 4: merge partials (LSE combine) ----------
__global__ __launch_bounds__(512) void combine(const float* __restrict__ part,
                                               float* __restrict__ ctx) {
  int t = blockIdx.x;                 // 448
  int bh = t / QTILES, qt = t % QTILES;
  int b = bh / NHEAD, h = bh % NHEAD;
  int row = threadIdx.x >> 5;         // 0..15
  int dim = threadIdx.x & 31;         // 0..31
  const float* rec0 = part + (size_t)t * NSEG * RECF;

  float M = -INFINITY;
#pragma unroll
  for (int s = 0; s < NSEG; ++s) M = fmaxf(M, rec0[s * RECF + 512 + row]);
  float L = 0.f, O = 0.f;
#pragma unroll
  for (int s = 0; s < NSEG; ++s) {
    const float* rec = rec0 + s * RECF;
    float sc = __expf(rec[512 + row] - M);
    L += sc * rec[528 + row];
    O += sc * rec[row * 32 + dim];
  }
  int qrow = qt * 16 + row;
  if (qrow < QLEN)
    ctx[((size_t)b * QLEN + qrow) * DMODEL + h * HDIM + dim] = O / L;
}

// ---------- kernel 5: out-proj + residual + LayerNorm ----------
__global__ __launch_bounds__(256) void out_ln(const float* __restrict__ ctx,
                                              const float* __restrict__ query,
                                              const float* __restrict__ Wout,
                                              const float* __restrict__ bout,
                                              const float* __restrict__ gamma,
                                              const float* __restrict__ beta,
                                              float* __restrict__ out) {
  int row = blockIdx.x;     // 800 rows
  int d = threadIdx.x;      // 256
  __shared__ float srow[DMODEL];
  __shared__ float red[DMODEL];
  __shared__ float s_mu, s_var;

  srow[d] = ctx[(size_t)row * DMODEL + d];
  __syncthreads();

  float acc = bout[d];
  const float* wr = Wout + (size_t)d * DMODEL;
#pragma unroll 4
  for (int k = 0; k < DMODEL; k += 4) {
    float4 w4 = *(const float4*)(wr + k);
    acc += srow[k] * w4.x + srow[k + 1] * w4.y + srow[k + 2] * w4.z +
           srow[k + 3] * w4.w;
  }
  float x = query[(size_t)row * DMODEL + d] + acc;

  red[d] = x; __syncthreads();
  for (int s = 128; s > 0; s >>= 1) { if (d < s) red[d] += red[d + s]; __syncthreads(); }
  if (d == 0) s_mu = red[0] * (1.f / DMODEL);
  __syncthreads();
  float xc = x - s_mu;
  red[d] = xc * xc; __syncthreads();
  for (int s = 128; s > 0; s >>= 1) { if (d < s) red[d] += red[d + s]; __syncthreads(); }
  if (d == 0) s_var = red[0] * (1.f / DMODEL);
  __syncthreads();
  out[(size_t)row * DMODEL + d] = xc * rsqrtf(s_var + 1e-5f) * gamma[d] + beta[d];
}

// ---------- launch ----------
extern "C" void kernel_launch(void* const* d_in, const int* in_sizes, int n_in,
                              void* d_out, int out_size, void* d_ws, size_t ws_size,
                              hipStream_t stream) {
  const float* query          = (const float*)d_in[0];
  const float* key_value      = (const float*)d_in[1];
  const unsigned char* mask   = (const unsigned char*)d_in[2];  // jax bool -> u8
  const float* in_proj_weight = (const float*)d_in[3];
  const float* in_proj_bias   = (const float*)d_in[4];
  const float* out_proj_w     = (const float*)d_in[5];
  const float* out_proj_b     = (const float*)d_in[6];
  const float* ln_w           = (const float*)d_in[7];
  const float* ln_b           = (const float*)d_in[8];
  float* out = (float*)d_out;

  char* ws = (char*)d_ws;
  size_t off = 0;
  __bf16* qbf  = (__bf16*)(ws + off); off += (size_t)BH * QPAD * HDIM * 2;       // 448 KiB
  __bf16* kbf  = (__bf16*)(ws + off); off += (size_t)BH * HW * HDIM * 2;         // 64 MiB
  __bf16* vT   = (__bf16*)(ws + off); off += (size_t)BH * HW * HDIM * 2;         // 64 MiB
  float*  part = (float*)(ws + off);  off += (size_t)NTILES * NSEG * RECF * 4;   // ~7.4 MiB
  float*  ctx  = (float*)(ws + off);  off += (size_t)BATCH * QLEN * DMODEL * 4;  // 800 KiB
  __bf16* wbf  = (__bf16*)(ws + off); off += (size_t)3 * DMODEL * DMODEL * 2;    // 384 KiB
  __bf16* kvbf = (__bf16*)(ws + off); off += (size_t)BATCH * HW * DMODEL * 2;    // 64 MiB

  // pre-convert weights (768x256) and key_value (8x16384x256) to bf16
  cvt_bf16<<<(3 * DMODEL * DMODEL / 4 + 255) / 256, 256, 0, stream>>>(
      in_proj_weight, wbf, 3 * DMODEL * DMODEL / 4);
  cvt_bf16<<<((int)((size_t)BATCH * HW * DMODEL / 4) + 255) / 256, 256, 0, stream>>>(
      key_value, kvbf, (int)((size_t)BATCH * HW * DMODEL / 4));

  kv_proj<<<4096, 256, 0, stream>>>(kvbf, wbf, in_proj_bias, kbf, vT);
  q_proj<<<28, 256, 0, stream>>>(query, wbf, in_proj_bias, qbf);
  attn_partial<<<NTILES, 256, 0, stream>>>(qbf, kbf, vT, mask, part);
  combine<<<NTILES, 512, 0, stream>>>(part, ctx);
  out_ln<<<BATCH * QLEN, 256, 0, stream>>>(ctx, query, out_proj_w, out_proj_b,
                                           ln_w, ln_b, out);
}
